// SCL_66632122630298
// MI455X (gfx1250) — matrix operand order repaired and checked by reading, not compiled, not run on hardware
//
#include <hip/hip_runtime.h>
#include <hip/hip_bf16.h>
#include <math.h>

typedef _Float16 v16h __attribute__((ext_vector_type(16)));
typedef _Float16 v8h  __attribute__((ext_vector_type(8)));
typedef _Float16 v4h  __attribute__((ext_vector_type(4)));
typedef float    v8f  __attribute__((ext_vector_type(8)));
typedef unsigned int v8u __attribute__((ext_vector_type(8)));

#define STR_H 264                       // activation row stride in halves (528B = 33*16B, bank-staggered)
#define X3S   68                        // x3^T column stride in floats (bank-conflict-free Householder reads)
#define WAVES 4
#define WAVE_LDS_F (16 * 8 * X3S)       // 8704 floats = 34816 B per wave (x3t; aliases 8448B activation buf)
// ws layout (dwords): W1 frags [0,8192), W2 frags [8192,40960), W3 frags [40960,106496)

// Branch-free tanh: hardware V_TANH_F32 if the builtin exists, else exp2/rcp sequence.
__device__ __forceinline__ float fast_tanh(float x) {
#if __has_builtin(__builtin_amdgcn_tanh_f32)
  return __builtin_amdgcn_tanh_f32(x);
#elif __has_builtin(__builtin_amdgcn_tanhf)
  return __builtin_amdgcn_tanhf(x);
#else
  const float ax = __builtin_fabsf(x);
  const float e  = __builtin_amdgcn_exp2f(ax * -2.8853900817779268f);  // exp(-2|x|)
  const float t  = (1.0f - e) * __builtin_amdgcn_rcpf(1.0f + e);
  return __builtin_copysignf(t, x);
#endif
}

// ---------------------------------------------------------------------------
// Weight pre-swizzle: f32 row-major (K x N) -> f16 WMMA B-fragment tiles.
// Tile (kt,nt) = 256 dwords; dword (lane,v) holds W[kt*32 + 16*(lane/16) + 2v .. +1][nt*16 + lane%16]
// ---------------------------------------------------------------------------
__global__ void __launch_bounds__(256) convert_weights_kernel(
    const float* __restrict__ W1, const float* __restrict__ W2,
    const float* __restrict__ W3, unsigned int* __restrict__ wf)
{
  const int tid = blockIdx.x * 256 + threadIdx.x;    // 0 .. 106495
  const float* W; int N; int base; int d;
  if (tid < 8192)       { W = W1; N = 256; base = 0;     d = tid; }
  else if (tid < 40960) { W = W2; N = 256; base = 8192;  d = tid - 8192; }
  else                  { W = W3; N = 512; base = 40960; d = tid - 40960; }
  const int chunk = d >> 8;
  const int r     = d & 255;
  const int lane  = r >> 3;
  const int v     = r & 7;
  const int nt    = chunk % (N >> 4);
  const int kt    = chunk / (N >> 4);
  const int k = kt * 32 + ((lane >> 4) << 4) + 2 * v;
  const int n = nt * 16 + (lane & 15);
  const _Float16 h0 = (_Float16)W[(size_t)k * N + n];
  const _Float16 h1 = (_Float16)W[(size_t)(k + 1) * N + n];
  const unsigned int lo = (unsigned int)__builtin_bit_cast(unsigned short, h0);
  const unsigned int hi = (unsigned int)__builtin_bit_cast(unsigned short, h1);
  wf[base + d] = lo | (hi << 16);
}

// A-fragment (16x32 f16) from LDS row-major activations, in contiguous k order (same as the B pre-swizzle):
// lane: VGPR0-3 hold K = kt*32 + (lane/16)*16 + 0..7 ; VGPR4-7 hold +8.  Two ds_load_b128.
__device__ __forceinline__ v16h load_afrag(const _Float16* xa, int mrow, int half, int kt) {
  const int k0 = kt * 32 + half * 16;
  const v8h lo = *(const v8h*)(xa + mrow * STR_H + k0);
  const v8h hi = *(const v8h*)(xa + mrow * STR_H + k0 + 8);
  v16h a;
#pragma unroll
  for (int i = 0; i < 8; ++i) { a[i] = lo[i]; a[i + 8] = hi[i]; }
  return a;
}

// B-fragment: 32 contiguous bytes per lane from pre-swizzled tile (two global_load_b128).
__device__ __forceinline__ v16h load_bfrag(const unsigned int* __restrict__ wf, int chunk, int lane) {
  const v8u w = *(const v8u*)(wf + (chunk << 8) + (lane << 3));
  return __builtin_bit_cast(v16h, w);
}

// ---------------------------------------------------------------------------
// Fused: MLP (3x WMMA GEMM, tanh) -> Householder(64x8) -> a* = M^T a -> q = M a*
// 4 waves/block, 16 batch rows per wave.
// ---------------------------------------------------------------------------
__global__ void __launch_bounds__(WAVES * 32) scl_fused_kernel(
    const float* __restrict__ cond, const float* __restrict__ act,
    const float* __restrict__ b1v,  const float* __restrict__ b2v,
    const float* __restrict__ b3v,
    const unsigned int* __restrict__ wf1, const unsigned int* __restrict__ wf2,
    const unsigned int* __restrict__ wf3, float* __restrict__ out)
{
  __shared__ __align__(16) float smem[WAVES * WAVE_LDS_F];   // 139264 B
  const int wave = threadIdx.x >> 5;
  const int lane = threadIdx.x & 31;
  const int half = lane >> 4;
  const int mrow = lane & 15;
  float*    x3t  = smem + wave * WAVE_LDS_F;                 // [16][8][X3S] f32 (GEMM3 output, transposed)
  _Float16* xa   = (_Float16*)x3t;                           // [16][STR_H] f16 activations (aliased, phase-ordered)
  const int row0 = (blockIdx.x * WAVES + wave) << 4;

  // ---- Stage cond (16x64 f32) -> xa f16 -------------------------------------------------
#pragma unroll
  for (int i = 0; i < 8; ++i) {
    const int e = i * 128 + lane * 4;
    const int r = e >> 6, c = e & 63;
    const float4 f = *(const float4*)(cond + (size_t)(row0 + r) * 64 + c);
    v4h h; h[0] = (_Float16)f.x; h[1] = (_Float16)f.y; h[2] = (_Float16)f.z; h[3] = (_Float16)f.w;
    *(v4h*)(xa + r * STR_H + c) = h;
  }
  __syncthreads();

  // ---- Layer 1: x1 = tanh(cond @ W1 + b1)   K=64 (2 kt), N=256 (16 nt) ------------------
  {
    const v16h a0 = load_afrag(xa, mrow, half, 0);
    const v16h a1 = load_afrag(xa, mrow, half, 1);
#pragma unroll 2
    for (int nt = 0; nt < 16; ++nt) {
      v8f c = {};
      c = __builtin_amdgcn_wmma_f32_16x16x32_f16(false, a0, false, load_bfrag(wf1, nt,      lane), (short)0, c, false, false);
      c = __builtin_amdgcn_wmma_f32_16x16x32_f16(false, a1, false, load_bfrag(wf1, 16 + nt, lane), (short)0, c, false, false);
      const float bb = b1v[nt * 16 + mrow];
#pragma unroll
      for (int g = 0; g < 8; ++g)
        xa[(g + 8 * half) * STR_H + nt * 16 + mrow] = (_Float16)fast_tanh(c[g] + bb);
    }
  }
  __syncthreads();

  // ---- Layer 2: x2 = tanh(x1 @ W2 + b2)   K=256 (8 kt), N=256 (16 nt) -------------------
  {
    v16h a[8];
#pragma unroll
    for (int kt = 0; kt < 8; ++kt) a[kt] = load_afrag(xa, mrow, half, kt);
#pragma unroll 2
    for (int nt = 0; nt < 16; ++nt) {
      v8f c = {};
#pragma unroll
      for (int kt = 0; kt < 8; ++kt)
        c = __builtin_amdgcn_wmma_f32_16x16x32_f16(false, a[kt], false, load_bfrag(wf2, kt * 16 + nt, lane), (short)0, c, false, false);
      const float bb = b2v[nt * 16 + mrow];
#pragma unroll
      for (int g = 0; g < 8; ++g)
        xa[(g + 8 * half) * STR_H + nt * 16 + mrow] = (_Float16)fast_tanh(c[g] + bb);
    }
  }
  __syncthreads();

  // ---- Layer 3: x3 = x2 @ W3 + b3   K=256 (8 kt), N=512 (32 nt) -> x3t[r][j%8][j/8] -----
  {
    v16h a[8];
#pragma unroll
    for (int kt = 0; kt < 8; ++kt) a[kt] = load_afrag(xa, mrow, half, kt);
#pragma unroll 2
    for (int nt = 0; nt < 32; ++nt) {
      v8f c = {};
#pragma unroll
      for (int kt = 0; kt < 8; ++kt)
        c = __builtin_amdgcn_wmma_f32_16x16x32_f16(false, a[kt], false, load_bfrag(wf3, kt * 32 + nt, lane), (short)0, c, false, false);
      const int j  = nt * 16 + mrow;       // 0..511
      const float bb = b3v[j];
      const int cc = j & 7, nn = j >> 3;
#pragma unroll
      for (int g = 0; g < 8; ++g)
        x3t[((g + 8 * half) * 8 + cc) * X3S + nn] = c[g] + bb;
    }
  }
  __syncthreads();

  // ---- Householder: M = prod_c (I - 2 v_c v_c^T / v^T v) [:, :8]; then a*, q -----------
#pragma unroll 1
  for (int r = 0; r < 16; ++r) {
    float M0[8], M1[8];                     // lane holds rows n=lane and n=lane+32 of M (64x8)
#pragma unroll
    for (int k = 0; k < 8; ++k) { M0[k] = (lane == k) ? 1.0f : 0.0f; M1[k] = 0.0f; }
#pragma unroll 1
    for (int c = 0; c < 8; ++c) {
      const float v0 = x3t[(r * 8 + c) * X3S + lane];
      const float v1 = x3t[(r * 8 + c) * X3S + lane + 32];
      float p[9];
#pragma unroll
      for (int k = 0; k < 8; ++k) p[k] = v0 * M0[k] + v1 * M1[k];   // v^T M
      p[8] = v0 * v0 + v1 * v1;                                      // v^T v
#pragma unroll
      for (int off = 16; off > 0; off >>= 1)
#pragma unroll
        for (int k = 0; k < 9; ++k) p[k] += __shfl_xor(p[k], off, 32);
      const float s = 2.0f / p[8];
#pragma unroll
      for (int k = 0; k < 8; ++k) { const float t = p[k] * s; M0[k] -= v0 * t; M1[k] -= v1 * t; }
    }
    const float a0 = act[(size_t)(row0 + r) * 64 + lane];
    const float a1 = act[(size_t)(row0 + r) * 64 + lane + 32];
    float q[8];
#pragma unroll
    for (int k = 0; k < 8; ++k) q[k] = a0 * M0[k] + a1 * M1[k];      // a* = M^T a
#pragma unroll
    for (int off = 16; off > 0; off >>= 1)
#pragma unroll
      for (int k = 0; k < 8; ++k) q[k] += __shfl_xor(q[k], off, 32);
    float o0 = 0.0f, o1 = 0.0f;
#pragma unroll
    for (int k = 0; k < 8; ++k) { o0 += M0[k] * q[k]; o1 += M1[k] * q[k]; }   // q = M a*
    out[(size_t)(row0 + r) * 64 + lane]      = o0;
    out[(size_t)(row0 + r) * 64 + lane + 32] = o1;
  }
}

extern "C" void kernel_launch(void* const* d_in, const int* in_sizes, int n_in,
                              void* d_out, int out_size, void* d_ws, size_t ws_size,
                              hipStream_t stream) {
  (void)in_sizes; (void)n_in; (void)out_size; (void)ws_size;
  const float* cond = (const float*)d_in[0];
  const float* act  = (const float*)d_in[1];
  const float* W1   = (const float*)d_in[2];
  const float* b1   = (const float*)d_in[3];
  const float* W2   = (const float*)d_in[4];
  const float* b2   = (const float*)d_in[5];
  const float* W3   = (const float*)d_in[6];
  const float* b3   = (const float*)d_in[7];
  unsigned int* wf  = (unsigned int*)d_ws;       // 106496 dwords = 416 KB

  convert_weights_kernel<<<416, 256, 0, stream>>>(W1, W2, W3, wf);
  scl_fused_kernel<<<8192 / (WAVES * 16), WAVES * 32, 0, stream>>>(
      cond, act, b1, b2, b3, wf, wf + 8192, wf + 40960, (float*)d_out);
}
